// RelativeMultiHeadAttention_27479200759858
// MI455X (gfx1250) — matrix operand order repaired and checked by reading, not compiled, not run on hardware
//
#include <hip/hip_runtime.h>
#include <hip/hip_bf16.h>
#include <stdint.h>

// ---------------------------------------------------------------------------
// Relative (Transformer-XL) multi-head attention for MI455X / gfx1250.
//  K0 cvt_kernel  : one-shot f32 -> bf16 conversion of activations + weights
//                   (GEMM operands then stream as bf16: half the HBM bytes,
//                   zero conversion VALU in the WMMA hot loops).
//  K1 proj_kernel : QU/QV/K/Vt/Prev projections, pure bf16 WMMA pipeline.
//  K2 attn_kernel : fused flash attention; K/P/V slabs staged to LDS via TDM
//                   (tensor_load_to_lds + s_wait_tensorcnt); exact rel-shift
//                   via WMMA-computed diagonal R-bands gathered from LDS.
//  K3 outproj     : context @ Wo^T + bo -> f32 output.
// ---------------------------------------------------------------------------

typedef __attribute__((ext_vector_type(16))) __bf16        v16bf;
typedef __attribute__((ext_vector_type(2)))  __bf16        v2bf;
typedef __attribute__((ext_vector_type(8)))  float         v8f;
typedef __attribute__((ext_vector_type(4)))  unsigned int  u32x4;
typedef __attribute__((ext_vector_type(8)))  int           i32x8;
typedef __attribute__((ext_vector_type(4)))  int           i32x4;

#define D_MODEL 512
#define NHEAD   16
#define DHEAD   32
#define BATCH   4
#define SEQ     1024

// Hardware bf16 conversion (clang picks v_cvt_pk_bf16_f32 on gfx1250).
__device__ __forceinline__ unsigned short bfbits(float x) {
  __bf16 h = (__bf16)x;
  return __builtin_bit_cast(unsigned short, h);
}
__device__ __forceinline__ unsigned pack2(float a, float b) {
  v2bf t; t[0] = (__bf16)a; t[1] = (__bf16)b;
  return __builtin_bit_cast(unsigned, t);
}
struct U32x8s { uint4 lo, hi; };
__device__ __forceinline__ v16bf mkv16(uint4 lo, uint4 hi) {
  U32x8s t{lo, hi};
  return __builtin_bit_cast(v16bf, t);
}
__device__ __forceinline__ v8f wmma_bf16(v16bf a, v16bf b, v8f c) {
  return __builtin_amdgcn_wmma_f32_16x16x32_bf16(false, a, false, b,
                                                 (short)0, c, false, false);
}
__device__ __forceinline__ float rmax16(float v) {
#pragma unroll
  for (int m = 1; m < 16; m <<= 1) v = fmaxf(v, __shfl_xor(v, m, 32));
  return v;  // butterfly stays inside 16-lane half (rows of the C tile)
}
__device__ __forceinline__ float rsum16(float v) {
#pragma unroll
  for (int m = 1; m < 16; m <<= 1) v += __shfl_xor(v, m, 32);
  return v;
}
__device__ __forceinline__ int imin(int a, int b) { return a < b ? a : b; }
__device__ __forceinline__ int imax(int a, int b) { return a > b ? a : b; }

// ---------------------------------------------------------------------------
// K0: bulk f32 -> bf16 (float4 in, 4x bf16 out per thread-iter)
// ---------------------------------------------------------------------------
__global__ __launch_bounds__(256) void cvt_kernel(
    const float* __restrict__ src, unsigned short* __restrict__ dst, int n4) {
  int i = blockIdx.x * blockDim.x + threadIdx.x;
  if (i < n4) {
    float4 f = ((const float4*)src)[i];
    uint2 o;
    o.x = pack2(f.x, f.y);
    o.y = pack2(f.z, f.w);
    ((uint2*)dst)[i] = o;
  }
}

// ---------------------------------------------------------------------------
// K1: fused projections.  z = 0:QU(q+u)  1:QV(q+v)  2:K  3:V(transposed)  4:P(reversed)
// One wave per 16x16 tile of the [4096 x 512] x [512 x 512]^T GEMM, K-loop of
// 16 x v_wmma_f32_16x16x32_bf16 fed by bf16 global_load_b128 only.
// A/B both in contiguous k order (lane<16: K kc..kc+15;
// lane>=16: kc+16..kc+31; A and B: 16 contiguous K at kc + half*16).
// ---------------------------------------------------------------------------
__global__ __launch_bounds__(256) void proj_kernel(
    const unsigned short* __restrict__ qb, const unsigned short* __restrict__ kb_,
    const unsigned short* __restrict__ vb_, const unsigned short* __restrict__ pb,
    const unsigned short* __restrict__ Wqb, const unsigned short* __restrict__ Wkb,
    const unsigned short* __restrict__ Wvb, const unsigned short* __restrict__ Wpb,
    const float* __restrict__ bq, const float* __restrict__ bk,
    const float* __restrict__ bv, const float* __restrict__ ub,
    const float* __restrict__ vb,
    unsigned short* __restrict__ QU, unsigned short* __restrict__ QV,
    unsigned short* __restrict__ KK, unsigned short* __restrict__ VT,
    unsigned short* __restrict__ PR) {
  const int lane = threadIdx.x & 31, wid = threadIdx.x >> 5;
  const int col16 = lane & 15, hi = lane >> 4;
  const int z = blockIdx.z;
  const unsigned short* X = nullptr; const unsigned short* W = nullptr;
  const float* bias = nullptr; const float* hb = nullptr;
  unsigned short* out = nullptr;
  switch (z) {
    case 0: X = qb;  W = Wqb; bias = bq; hb = ub; out = QU; break;
    case 1: X = qb;  W = Wqb; bias = bq; hb = vb; out = QV; break;
    case 2: X = kb_; W = Wkb; bias = bk;          out = KK; break;
    case 3: X = vb_; W = Wvb; bias = bv;          out = VT; break;
    default:X = pb;  W = Wpb;                     out = PR; break;
  }
  const int m0 = (blockIdx.x * 8 + wid) * 16;
  const int n0 = blockIdx.y * 16;
  const int o  = n0 + col16;                              // feature (h*32+d)
  const unsigned short* Xr = X + (size_t)(m0 + col16) * D_MODEL;
  const unsigned short* Wr = W + (size_t)o * D_MODEL;

  v8f c = {};
#pragma unroll 4
  for (int kc = 0; kc < D_MODEL; kc += 32) {
    const int ka = kc + (hi ? 16 : 0);                    // A: 16 contiguous
    v16bf A = mkv16(*(const uint4*)(Xr + ka), *(const uint4*)(Xr + ka + 8));
    const int kb = kc + (hi ? 16 : 0);                    // B: 16 contiguous
    v16bf Bv = mkv16(*(const uint4*)(Wr + kb), *(const uint4*)(Wr + kb + 8));
    c = wmma_bf16(A, Bv, c);
  }
  float badd = 0.f;
  if (bias) badd += bias[o];
  if (hb)   badd += hb[o];                                // u/v bias folded in
  const int h = o >> 5, d = o & 31;
#pragma unroll
  for (int r = 0; r < 8; ++r) {
    int m = m0 + r + hi * 8;                              // C rows per half
    int bi = m >> 10, s = m & 1023;
    unsigned short val = bfbits(c[r] + badd);
    size_t idx;
    if (z == 3)       idx = (((size_t)(bi * NHEAD + h)) * DHEAD + d) * SEQ + s;             // V: [B,H,D,S]
    else if (z == 4)  idx = (((size_t)(bi * NHEAD + h)) * SEQ + (SEQ - 1 - s)) * DHEAD + d; // P reversed
    else              idx = (((size_t)(bi * NHEAD + h)) * SEQ + s) * DHEAD + d;             // [B,H,S,D]
    out[idx] = val;
  }
}

// ---------------------------------------------------------------------------
// K2: fused flash attention with exact relative shift.
// ---------------------------------------------------------------------------
#define LDS_K        0
#define LDS_P        65536
#define LDS_V        131072
#define LDS_WAVE     196608
#define WAVE_SCRATCH 5120   // band1(2KB f32) + band2(2KB f32) + probs(1KB bf16)

#if __has_builtin(__builtin_amdgcn_tensor_load_to_lds)
#define HAVE_TDM 1
#else
#define HAVE_TDM 0
#endif

#if HAVE_TDM
__device__ __forceinline__ void tdm_stage(unsigned ldsAddr, const void* g,
                                          unsigned bytes) {
  // 1-D tensor descriptor (D#) per CDNA5 ISA:
  // data_size=8B, tensor_dim0=tile_dim0=bytes/8, tensor_dim1=tile_dim1=1.
  unsigned long long ga = (unsigned long long)g;
  unsigned n8 = bytes >> 3;
  u32x4 g0; i32x8 g1;
  g0[0] = 1u;                                                  // count=1
  g0[1] = ldsAddr;                                             // lds_addr
  g0[2] = (unsigned)ga;                                        // gaddr[31:0]
  g0[3] = (unsigned)((ga >> 32) & 0x01FFFFFFull) | (2u << 30); // [56:32]|type=2
  g1[0] = 0x00030000;                                 // wg_mask=0, data_size=8B
  g1[1] = (int)((n8 & 0xFFFFu) << 16);                // tensor_dim0[15:0]
  g1[2] = (int)(((n8 >> 16) & 0xFFFFu) | (1u << 16)); // dim0[31:16] | dim1=1
  g1[3] = (int)((n8 & 0xFFFFu) << 16);                // tile_dim0
  g1[4] = 1;                                          // tile_dim1=1
  g1[5] = (int)n8;                                    // tensor_dim0_stride
  g1[6] = 0; g1[7] = 0;
  i32x4 z4 = {};
#if defined(__clang_major__) && __clang_major__ >= 23
  i32x8 z8 = {};
  __builtin_amdgcn_tensor_load_to_lds(g0, g1, z4, z4, z8, 0);
#else
  __builtin_amdgcn_tensor_load_to_lds(g0, g1, z4, z4, 0);
#endif
}
#endif

__global__ __launch_bounds__(256) void attn_kernel(
    const unsigned short* __restrict__ QU, const unsigned short* __restrict__ QV,
    const unsigned short* __restrict__ KK, const unsigned short* __restrict__ VT,
    const unsigned short* __restrict__ PR, unsigned short* __restrict__ CTX) {
  extern __shared__ char smem[];
  unsigned short* Ks = (unsigned short*)(smem + LDS_K);  // [t][d] bf16, 64KB
  unsigned short* Ps = (unsigned short*)(smem + LDS_P);  // [r][d] reversed-p
  unsigned short* Vs = (unsigned short*)(smem + LDS_V);  // [d][t] transposed
  const int bh = blockIdx.x;                             // (b,h)
  const size_t slab = (size_t)bh * (SEQ * DHEAD);

#if HAVE_TDM
  if (threadIdx.x < 32) {                                // one wave issues TDM
    unsigned base = (unsigned)(unsigned long long)(void*)smem; // LDS byte addr
    tdm_stage(base + LDS_K, KK + slab, SEQ * DHEAD * 2);
    tdm_stage(base + LDS_P, PR + slab, SEQ * DHEAD * 2);
    tdm_stage(base + LDS_V, VT + slab, SEQ * DHEAD * 2);
    __builtin_amdgcn_s_wait_tensorcnt(0);                // s_wait_tensorcnt 0
  }
#else
  {
    const uint4* gk = (const uint4*)(KK + slab);
    const uint4* gp = (const uint4*)(PR + slab);
    const uint4* gv = (const uint4*)(VT + slab);
    uint4* lk = (uint4*)Ks; uint4* lp = (uint4*)Ps; uint4* lv = (uint4*)Vs;
    for (int i = threadIdx.x; i < SEQ * DHEAD / 8; i += blockDim.x) {
      lk[i] = gk[i]; lp[i] = gp[i]; lv[i] = gv[i];
    }
  }
#endif
  __syncthreads();

  const int lane = threadIdx.x & 31, wid = threadIdx.x >> 5;
  const int col16 = lane & 15, hi = lane >> 4;
  float* band1 = (float*)(smem + LDS_WAVE + wid * WAVE_SCRATCH); // R[i, i-j]
  float* band2 = band1 + 512;                                    // R[i+1, ...]
  unsigned short* probs = (unsigned short*)(band2 + 512);        // 16x32 bf16
  const int bi = bh >> 4, h = bh & 15;
  const float rs = 0.04419417382415922f;  // 1/sqrt(d_model)

  for (int it = blockIdx.y * 8 + wid; it < SEQ / 16; it += 16) {
    const int i0 = it * 16;
    auto loadA = [&](const unsigned short* base, int row) -> v16bf {
      const unsigned short* p = base + slab + (size_t)row * DHEAD + (hi ? 16 : 0);
      return mkv16(*(const uint4*)p, *(const uint4*)(p + 8));
    };
    v16bf Aqu = loadA(QU, i0 + col16);                       // rows i0..i0+15
    v16bf Aqv = loadA(QV, i0 + col16);
    v16bf Aq2 = loadA(QV, imin(i0 + 1 + col16, SEQ - 1));    // rows i0+1..i0+16
    v8f m8, l8, acc0 = {}, acc1 = {};
#pragma unroll
    for (int r = 0; r < 8; ++r) { m8[r] = -1e30f; l8[r] = 0.f; }

#pragma unroll 1
    for (int jc = 0; jc < SEQ; jc += 32) {
      v8f cs0, cs1;
#pragma unroll
      for (int sub = 0; sub < 2; ++sub) {
        const int j0c = jc + sub * 16;
        v8f zero = {};
        // content score: (q+u) . k^T
        v16bf Bk = *(const v16bf*)(const void*)&Ks[(j0c + col16) * DHEAD + hi * 16];
        v8f c = wmma_bf16(Aqu, Bk, zero);
        // R-bands for the relative shift: R = (q+v) . p_rev^T
        const int rb1 = i0 - j0c - 15;        // band1 covers r in [rb1, rb1+31]
        auto ldP = [&](int r_) -> v16bf {
          r_ = imin(imax(r_, 0), SEQ - 1);    // clamp: OOB cols never gathered
          return *(const v16bf*)(const void*)&Ps[r_ * DHEAD + hi * 16];
        };
        v8f b1a = wmma_bf16(Aqv, ldP(rb1 + col16), zero);
        v8f b1b = wmma_bf16(Aqv, ldP(rb1 + 16 + col16), zero);
        const int rb2 = rb1 + SEQ + 1;        // band2 covers r' in [rb2, rb2+31]
        v8f b2a = wmma_bf16(Aq2, ldP(rb2 + col16), zero);
        v8f b2b = wmma_bf16(Aq2, ldP(rb2 + 16 + col16), zero);
#pragma unroll
        for (int r = 0; r < 8; ++r) {         // spill bands to LDS scratch
          int rl = r + hi * 8;
          band1[rl * 32 + col16]      = b1a[r];
          band1[rl * 32 + 16 + col16] = b1b[r];
          band2[rl * 32 + col16]      = b2a[r];
          band2[rl * 32 + 16 + col16] = b2b[r];
        }
        v8f sc;
#pragma unroll
        for (int r = 0; r < 8; ++r) {         // exact _relative_shift gather
          int rl = r + hi * 8;
          int iG = i0 + rl, jG = j0c + col16;
          int gc = rl - col16 + 15;           // == (i-j)-rb1 == (S+1+i-j)-rb2
          float pos = (jG <= iG)     ? band1[rl * 32 + gc]
                      : (jG == iG + 1) ? 0.f
                                       : band2[rl * 32 + gc];
          sc[r] = (c[r] + pos) * rs;
        }
        if (sub == 0) cs0 = sc; else cs1 = sc;
      }
      // online softmax over this 32-wide block (16-lane row butterflies)
#pragma unroll
      for (int r = 0; r < 8; ++r) {
        int rl = r + hi * 8;
        float mx    = rmax16(fmaxf(cs0[r], cs1[r]));
        float mnew  = fmaxf(m8[r], mx);
        float scale = __expf(m8[r] - mnew);
        float e0 = __expf(cs0[r] - mnew);
        float e1 = __expf(cs1[r] - mnew);
        l8[r] = l8[r] * scale + rsum16(e0 + e1);
        m8[r] = mnew;
        acc0[r] *= scale; acc1[r] *= scale;
        probs[rl * 32 + col16]      = bfbits(e0);  // C-layout -> A-layout via LDS
        probs[rl * 32 + 16 + col16] = bfbits(e1);
      }
      // attn . V for this 32-column block (K=32 -> one WMMA per 16-d half)
      const unsigned short* pr = probs + col16 * 32 + (hi ? 16 : 0);
      v16bf Ap  = mkv16(*(const uint4*)pr, *(const uint4*)(pr + 8));
      v16bf Bv0 = *(const v16bf*)(const void*)&Vs[col16 * SEQ + jc + hi * 16];
      v16bf Bv1 = *(const v16bf*)(const void*)&Vs[(16 + col16) * SEQ + jc + hi * 16];
      acc0 = wmma_bf16(Ap, Bv0, acc0);
      acc1 = wmma_bf16(Ap, Bv1, acc1);
    }
    // epilogue: context[b, s, h*32+d] = acc / l   (bf16 for the out-proj GEMM)
#pragma unroll
    for (int r = 0; r < 8; ++r) {
      int iG = i0 + r + hi * 8;
      float inv = 1.f / l8[r];
      size_t base = ((size_t)(bi * SEQ + iG)) * D_MODEL + h * DHEAD;
      CTX[base + col16]      = bfbits(acc0[r] * inv);
      CTX[base + 16 + col16] = bfbits(acc1[r] * inv);
    }
  }
}

// ---------------------------------------------------------------------------
// K3: output projection: context @ Wo^T + bo -> f32
// ---------------------------------------------------------------------------
__global__ __launch_bounds__(256) void outproj_kernel(
    const unsigned short* __restrict__ CTX, const unsigned short* __restrict__ Wob,
    const float* __restrict__ bo, float* __restrict__ out) {
  const int lane = threadIdx.x & 31, wid = threadIdx.x >> 5;
  const int col16 = lane & 15, hi = lane >> 4;
  const int m0 = (blockIdx.x * 8 + wid) * 16;
  const int n0 = blockIdx.y * 16;
  const int o  = n0 + col16;
  const unsigned short* Xr = CTX + (size_t)(m0 + col16) * D_MODEL;
  const unsigned short* Wr = Wob + (size_t)o * D_MODEL;
  v8f c = {};
#pragma unroll 4
  for (int kc = 0; kc < D_MODEL; kc += 32) {
    const int ka = kc + (hi ? 16 : 0);
    v16bf A  = mkv16(*(const uint4*)(Xr + ka), *(const uint4*)(Xr + ka + 8));
    const int kb = kc + (hi ? 16 : 0);
    v16bf Bv = mkv16(*(const uint4*)(Wr + kb), *(const uint4*)(Wr + kb + 8));
    c = wmma_bf16(A, Bv, c);
  }
  float badd = bo[o];
#pragma unroll
  for (int r = 0; r < 8; ++r) {
    int m = m0 + r + hi * 8;
    out[(size_t)m * D_MODEL + o] = c[r] + badd;
  }
}

// ---------------------------------------------------------------------------
extern "C" void kernel_launch(void* const* d_in, const int* in_sizes, int n_in,
                              void* d_out, int out_size, void* d_ws,
                              size_t ws_size, hipStream_t stream) {
  (void)in_sizes; (void)n_in; (void)out_size; (void)ws_size;
  const float* q  = (const float*)d_in[0];
  const float* k  = (const float*)d_in[1];
  const float* v  = (const float*)d_in[2];
  const float* pe = (const float*)d_in[3];
  // d_in[4] = mask: reference discards masked_fill result -> faithfully unused
  const float* Wq = (const float*)d_in[5];
  const float* bq = (const float*)d_in[6];
  const float* Wk = (const float*)d_in[7];
  const float* bk = (const float*)d_in[8];
  const float* Wv = (const float*)d_in[9];
  const float* bv = (const float*)d_in[10];
  const float* Wp = (const float*)d_in[11];
  const float* ub = (const float*)d_in[12];
  const float* vb = (const float*)d_in[13];
  const float* Wo = (const float*)d_in[14];
  const float* bo = (const float*)d_in[15];

  char* ws = (char*)d_ws;
  const size_t ACT  = (size_t)BATCH * SEQ * D_MODEL * 2;   // 4 MB bf16
  const size_t WGT  = (size_t)D_MODEL * D_MODEL * 2;       // 512 KB bf16
  unsigned short* Qb  = (unsigned short*)(ws);
  unsigned short* Kb  = (unsigned short*)(ws + 1 * ACT);
  unsigned short* Vb  = (unsigned short*)(ws + 2 * ACT);
  unsigned short* Pb  = (unsigned short*)(ws + 3 * ACT);
  char* wp = ws + 4 * ACT;
  unsigned short* Wqb = (unsigned short*)(wp + 0 * WGT);
  unsigned short* Wkb = (unsigned short*)(wp + 1 * WGT);
  unsigned short* Wvb = (unsigned short*)(wp + 2 * WGT);
  unsigned short* Wpb = (unsigned short*)(wp + 3 * WGT);
  unsigned short* Wob = (unsigned short*)(wp + 4 * WGT);
  char* sp = wp + 5 * WGT;
  unsigned short* QU  = (unsigned short*)(sp + 0 * ACT);
  unsigned short* QV  = (unsigned short*)(sp + 1 * ACT);
  unsigned short* KK  = (unsigned short*)(sp + 2 * ACT);
  unsigned short* VT  = (unsigned short*)(sp + 3 * ACT);
  unsigned short* PR  = (unsigned short*)(sp + 4 * ACT);
  unsigned short* CTX = (unsigned short*)(sp + 5 * ACT);   // ~42.5 MB total

  // K0: one-shot conversions (activations + weights)
  const int nAct4 = BATCH * SEQ * D_MODEL / 4;   // 524288
  const int nWgt4 = D_MODEL * D_MODEL / 4;       // 65536
  cvt_kernel<<<dim3((nAct4 + 255) / 256), 256, 0, stream>>>(q,  Qb,  nAct4);
  cvt_kernel<<<dim3((nAct4 + 255) / 256), 256, 0, stream>>>(k,  Kb,  nAct4);
  cvt_kernel<<<dim3((nAct4 + 255) / 256), 256, 0, stream>>>(v,  Vb,  nAct4);
  cvt_kernel<<<dim3((nAct4 + 255) / 256), 256, 0, stream>>>(pe, Pb,  nAct4);
  cvt_kernel<<<dim3((nWgt4 + 255) / 256), 256, 0, stream>>>(Wq, Wqb, nWgt4);
  cvt_kernel<<<dim3((nWgt4 + 255) / 256), 256, 0, stream>>>(Wk, Wkb, nWgt4);
  cvt_kernel<<<dim3((nWgt4 + 255) / 256), 256, 0, stream>>>(Wv, Wvb, nWgt4);
  cvt_kernel<<<dim3((nWgt4 + 255) / 256), 256, 0, stream>>>(Wp, Wpb, nWgt4);
  cvt_kernel<<<dim3((nWgt4 + 255) / 256), 256, 0, stream>>>(Wo, Wob, nWgt4);

  proj_kernel<<<dim3(32, 32, 5), 256, 0, stream>>>(
      Qb, Kb, Vb, Pb, Wqb, Wkb, Wvb, Wpb, bq, bk, bv, ub, vb,
      QU, QV, KK, VT, PR);

  // 192KB slab staging + 8 waves x 5KB scratch = 232KB dynamic LDS (<320KB/WGP)
  const size_t smembytes = LDS_WAVE + 8 * WAVE_SCRATCH;
  attn_kernel<<<dim3(BATCH * NHEAD, 2), 256, smembytes, stream>>>(
      QU, QV, KK, VT, PR, CTX);

  outproj_kernel<<<dim3(32, 32), 256, 0, stream>>>(CTX, Wob, bo, (float*)d_out);
}
